// SelfAttention_32925219291570
// MI455X (gfx1250) — matrix operand order repaired and checked by reading, not compiled, not run on hardware
//
#include <hip/hip_runtime.h>
#include <cmath>

#define WH   4096
#define NTOK 4280
#define NPAD 4288
#define MP   (2 * NPAD)
#define QD   320
#define NH   8
#define DH   40

typedef __attribute__((ext_vector_type(16))) __bf16 v16bf;
typedef __attribute__((ext_vector_type(8)))  __bf16 v8bf;
typedef __attribute__((ext_vector_type(8)))  float  v8f;

__device__ __forceinline__ v8f wmma_bf16(v16bf a, v16bf b, v8f c) {
  // D = A(16x32 bf16) * B(32x16 bf16) + C(16x16 f32)
  return __builtin_amdgcn_wmma_f32_16x16x32_bf16(false, a, false, b, (short)0, c,
                                                 false, false);
}

// A-matrix fragment (16x32 bf16). Lane l = row (l&15); elems = K kb..kb+15,
// kb = (l<16) ? 0 : 16.
__device__ __forceinline__ v16bf a_frag(const __bf16* row, int k0, int lane) {
  int kb = k0 + ((lane & 16) ? 16 : 0);
  union { v16bf v; v8bf h[2]; } u;
  u.h[0] = *(const v8bf*)(row + kb);
  u.h[1] = *(const v8bf*)(row + kb + 8);
  return u.v;
}

// B-matrix fragment (32x16 bf16) from B^T storage (rows = output column).
// Lane l = column (l&15); elems = K kb..kb+15, kb = (l<16) ? 0 : 16.
__device__ __forceinline__ v16bf bt_frag(const __bf16* btrow, int k0, int lane) {
  int kb = k0 + ((lane & 16) ? 16 : 0);
  return *(const v16bf*)(btrow + kb);
}

__device__ __forceinline__ float rmax16(float v) {
  v = fmaxf(v, __shfl_xor(v, 1)); v = fmaxf(v, __shfl_xor(v, 2));
  v = fmaxf(v, __shfl_xor(v, 4)); v = fmaxf(v, __shfl_xor(v, 8));
  return v;
}
__device__ __forceinline__ float rsum16(float v) {
  v += __shfl_xor(v, 1); v += __shfl_xor(v, 2);
  v += __shfl_xor(v, 4); v += __shfl_xor(v, 8);
  return v;
}

// ---------------- per-pixel object bitmask --------------------------------
__global__ void k_bits(const float* __restrict__ am, unsigned* __restrict__ bits) {
  int t = blockIdx.x * blockDim.x + threadIdx.x;
  if (t >= 2 * WH) return;
  int b = t >> 12, i = t & (WH - 1);
  unsigned v = 0;
#pragma unroll 1
  for (int o = 0; o < 30; ++o)
    v |= (am[((size_t)(b * 30 + o) << 12) | i] != 0.f) ? (1u << o) : 0u;
  bits[t] = v;
}

// mask(b,i,j) > 0 ?  (reconstructed from reference _build_mask)
__device__ __forceinline__ bool unmasked(int i, int j, unsigned rb, unsigned cb) {
  if (j >= NTOK) return false;          // padded keys
  if (i == j) return true;              // + eye * 1e-9
  if (i < WH) {
    if (j < WH) return (rb & cb) != 0;  // visual-visual: shared object
    int r = j - WH;
    if (r >= 120 || (r >= 30 && r < 90)) return true;
    return (rb >> (r % 30)) & 1;        // rep_cols: m[o][i]
  } else {
    if (j >= WH) return true;
    int r = i - WH;
    if (r >= 120 || (r >= 30 && r < 90)) return true;
    return (cb >> (r % 30)) & 1;        // rep_rows: m[o][j]
  }
}

// ---------------- fp32 -> bf16 converters ---------------------------------
__global__ void k_conv_x(const float* __restrict__ x, __bf16* __restrict__ xb) {
  int t = blockIdx.x * blockDim.x + threadIdx.x;
  if (t >= 2 * NTOK * QD) return;
  int c = t % QD, n = (t / QD) % NTOK, b = t / (QD * NTOK);
  xb[((size_t)(b * NPAD + n)) * QD + c] = (__bf16)x[t];
}
__global__ void k_conv_wt(const float* __restrict__ w, __bf16* __restrict__ wt) {
  int t = blockIdx.x * blockDim.x + threadIdx.x;
  if (t >= QD * QD) return;
  int j = t % QD, k = t / QD;
  wt[(size_t)j * QD + k] = (__bf16)w[t];   // transpose: wt[j][k] = w[k][j]
}

// ---------------- QKV projection (WMMA GEMM) ------------------------------
__global__ void __launch_bounds__(128)
k_proj(const __bf16* __restrict__ xb,
       const __bf16* __restrict__ wtq, const __bf16* __restrict__ wtk,
       const __bf16* __restrict__ wtv,
       __bf16* __restrict__ Q, __bf16* __restrict__ K, __bf16* __restrict__ VT) {
  const int tilesM = MP / 16, tilesN = QD / 16;
  int wave = (blockIdx.x * blockDim.x + threadIdx.x) >> 5;
  int lane = threadIdx.x & 31;
  int p = wave / (tilesM * tilesN);
  if (p >= 3) return;
  int r = wave % (tilesM * tilesN);
  int i0 = (r / tilesN) * 16, j0 = (r % tilesN) * 16;
  const __bf16* wt = (p == 0) ? wtq : (p == 1) ? wtk : wtv;

  int col = j0 + (lane & 15);
  const __bf16* arow  = xb + (size_t)(i0 + (lane & 15)) * QD;
  const __bf16* btrow = wt + (size_t)col * QD;
  v8f c = {};
#pragma unroll
  for (int k0 = 0; k0 < QD; k0 += 32)
    c = wmma_bf16(a_frag(arow, k0, lane), bt_frag(btrow, k0, lane), c);

  int b = (i0 >= NPAD), n0 = i0 - b * NPAD;
  int h = col / DH, d = col % DH;
  int grp8 = (lane & 16) ? 8 : 0;
#pragma unroll
  for (int e = 0; e < 8; ++e) {
    int n = n0 + e + grp8;
    __bf16 val = (__bf16)c[e];
    if (p == 2)
      VT[(((size_t)(b * NH + h) * 48) + d) * NPAD + n] = val;        // V^T[d][n]
    else
      (p == 0 ? Q : K)[(((size_t)(b * NH + h) * NPAD) + n) * 64 + d] = val;
  }
}

// ---------------- flash attention core ------------------------------------
// Block = 4 waves of the SAME (b,h), 4 consecutive 16-row query blocks.
// K/V chunks staged cooperatively into LDS via async-to-LDS, double buffered.
__global__ void __launch_bounds__(128)
k_attn(const __bf16* __restrict__ Q, const __bf16* __restrict__ K,
       const __bf16* __restrict__ VT, const unsigned* __restrict__ bits,
       __bf16* __restrict__ AO) {
  __shared__ alignas(32) __bf16 Ksh[2][32][64];   // 8 KB: 32 keys x 64 dims
  __shared__ alignas(32) __bf16 Vsh[2][64][32];   // 8 KB: 48(+pad) dims x 32 keys
  __shared__ alignas(32) __bf16 psh[4][16][32];   // 4 KB: per-wave P bounce

  int tid = threadIdx.x;
  int wv = tid >> 5, lane = tid & 31;
  const int BPB = (NPAD / 16) / 4;                // 67 blocks per (b,h)
  int bh = blockIdx.x / BPB;
  int i0 = ((blockIdx.x % BPB) * 4 + wv) * 16;
  int b = bh >> 3, h = bh & 7;

  const __bf16* Qb = Q + (size_t)bh * NPAD * 64;
  const char*   Kg = (const char*)(K + (size_t)bh * NPAD * 64);
  const char*   Vg = (const char*)(VT + (size_t)bh * 48 * NPAD);
  const unsigned* bb = bits + b * WH;
  __bf16 (*shp)[32] = psh[wv];

  // issue one chunk's cooperative async copies: 4 x B128 per lane (uniform)
  auto issue = [&](int j0, int buf) {
    unsigned long long gk = (unsigned long long)(Kg + (size_t)j0 * 128 + tid * 32);
    unsigned lk = (unsigned)(size_t)&Ksh[buf][0][0] + tid * 32;
    asm volatile("global_load_async_to_lds_b128 %0, %1, off"
                 :: "v"(lk), "v"(gk) : "memory");
    asm volatile("global_load_async_to_lds_b128 %0, %1, off offset:16"
                 :: "v"(lk), "v"(gk) : "memory");
#pragma unroll
    for (int s = 0; s < 2; ++s) {                 // 64 rows x 4 segs = 256 xfers
      int tr = 2 * tid + s;
      int row = tr >> 2, seg = tr & 3;
      unsigned long long gv =
          (unsigned long long)(Vg + ((size_t)row * NPAD + j0) * 2 + seg * 16);
      unsigned lv = (unsigned)(size_t)&Vsh[buf][row][0] + seg * 16;
      asm volatile("global_load_async_to_lds_b128 %0, %1, off"
                   :: "v"(lv), "v"(gv) : "memory");
    }
  };

  int grp8 = (lane & 16) ? 8 : 0;
  const __bf16* qrow = Qb + (size_t)(i0 + (lane & 15)) * 64;
  v16bf aq0 = a_frag(qrow, 0, lane);              // K-dim 0..31
  v16bf aq1 = a_frag(qrow, 32, lane);             // K-dim 32..63 (40..63 zero)

  int irow[8]; unsigned rbits[8];
  float mrow[8], lrow[8];
  v8f o0 = {}, o1 = {}, o2 = {};
#pragma unroll
  for (int e = 0; e < 8; ++e) {
    int i = i0 + e + grp8;
    irow[e] = i;
    rbits[e] = (i < WH) ? bb[i] : 0u;
    mrow[e] = -__builtin_inff();
    lrow[e] = 0.f;
  }
  const float scale = rsqrtf((float)DH);

  issue(0, 0);
  for (int j0 = 0; j0 < NPAD; j0 += 32) {
    int buf = (j0 >> 5) & 1;
    if (j0 + 32 < NPAD) {
      issue(j0 + 32, buf ^ 1);
      // async loads retire in order: <=4 outstanding => this chunk's 4 are done
      asm volatile("s_wait_asynccnt 0x4" ::: "memory");
    } else {
      asm volatile("s_wait_asynccnt 0x0" ::: "memory");
    }
    __syncthreads();                               // all waves' copies landed

    v8f s0 = {}, s1 = {};
    {
      const __bf16* kr0 = &Ksh[buf][lane & 15][0];
      const __bf16* kr1 = &Ksh[buf][16 + (lane & 15)][0];
      s0 = wmma_bf16(aq0, bt_frag(kr0, 0, lane), s0);
      s0 = wmma_bf16(aq1, bt_frag(kr0, 32, lane), s0);
      s1 = wmma_bf16(aq0, bt_frag(kr1, 0, lane), s1);
      s1 = wmma_bf16(aq1, bt_frag(kr1, 32, lane), s1);
    }
    int ja = j0 + (lane & 15), jb = ja + 16;
    unsigned ca = (ja < WH) ? bb[ja] : 0u;
    unsigned cb = (jb < WH) ? bb[jb] : 0u;
#pragma unroll
    for (int e = 0; e < 8; ++e) {
      float va = s0[e] * scale, vb = s1[e] * scale;
      s0[e] = unmasked(irow[e], ja, rbits[e], ca) ? va : -__builtin_inff();
      s1[e] = unmasked(irow[e], jb, rbits[e], cb) ? vb : -__builtin_inff();
    }
#pragma unroll
    for (int e = 0; e < 8; ++e) {
      float tm = rmax16(fmaxf(s0[e], s1[e]));
      float nm = fmaxf(fmaxf(mrow[e], tm), -1e30f);   // all-masked safe
      float cf = __expf(mrow[e] - nm);
      float p0 = __expf(s0[e] - nm);
      float p1 = __expf(s1[e] - nm);
      lrow[e] = lrow[e] * cf + rsum16(p0 + p1);
      mrow[e] = nm;
      o0[e] *= cf; o1[e] *= cf; o2[e] *= cf;
      shp[e + grp8][lane & 15]        = (__bf16)p0;   // C-layout -> LDS
      shp[e + grp8][16 + (lane & 15)] = (__bf16)p1;
    }
    asm volatile("s_wait_dscnt 0" ::: "memory");      // CDNA5 split DS counter
    v16bf pf = a_frag(&shp[lane & 15][0], 0, lane);   // reload in A-layout
    {
      o0 = wmma_bf16(pf, bt_frag(&Vsh[buf][(lane & 15)][0], 0, lane), o0);
      o1 = wmma_bf16(pf, bt_frag(&Vsh[buf][16 + (lane & 15)][0], 0, lane), o1);
      o2 = wmma_bf16(pf, bt_frag(&Vsh[buf][32 + (lane & 15)][0], 0, lane), o2);
    }
    __syncthreads();   // everyone done with buf before it is re-filled (j0+64)
  }

  int n = lane & 15;
#pragma unroll
  for (int e = 0; e < 8; ++e) {
    float inv = 1.f / lrow[e];
    size_t base = ((size_t)(b * NPAD + irow[e])) * QD + h * DH;
    AO[base + n]      = (__bf16)(o0[e] * inv);
    AO[base + 16 + n] = (__bf16)(o1[e] * inv);
    if (n < 8) AO[base + 32 + n] = (__bf16)(o2[e] * inv);
  }
}

// ---------------- output projection + bias --------------------------------
__global__ void __launch_bounds__(128)
k_out(const __bf16* __restrict__ AO, const __bf16* __restrict__ wto,
      const float* __restrict__ bo, float* __restrict__ out) {
  const int tilesM = MP / 16, tilesN = QD / 16;
  int wave = (blockIdx.x * blockDim.x + threadIdx.x) >> 5;
  int lane = threadIdx.x & 31;
  if (wave >= tilesM * tilesN) return;
  int i0 = (wave / tilesN) * 16, j0 = (wave % tilesN) * 16;
  int col = j0 + (lane & 15);
  const __bf16* arow  = AO + (size_t)(i0 + (lane & 15)) * QD;
  const __bf16* btrow = wto + (size_t)col * QD;
  v8f c = {};
#pragma unroll
  for (int k0 = 0; k0 < QD; k0 += 32)
    c = wmma_bf16(a_frag(arow, k0, lane), bt_frag(btrow, k0, lane), c);
  float bias = bo[col];
  int grp8 = (lane & 16) ? 8 : 0;
#pragma unroll
  for (int e = 0; e < 8; ++e) {
    int row = i0 + e + grp8;
    int b = (row >= NPAD), n = row - b * NPAD;
    if (n < NTOK)
      out[((size_t)(b * NTOK + n)) * QD + col] = c[e] + bias;
  }
}

// ---------------- host glue -----------------------------------------------
extern "C" void kernel_launch(void* const* d_in, const int* in_sizes, int n_in,
                              void* d_out, int out_size, void* d_ws, size_t ws_size,
                              hipStream_t stream) {
  const float* x  = (const float*)d_in[0];
  const float* am = (const float*)d_in[1];
  const float* Wq = (const float*)d_in[2];
  const float* Wk = (const float*)d_in[3];
  const float* Wv = (const float*)d_in[4];
  const float* Wo = (const float*)d_in[5];
  const float* bo = (const float*)d_in[6];
  float* out = (float*)d_out;

  char* base = (char*)d_ws;
  size_t off = 0;
  auto alloc = [&](size_t bytes) -> void* {
    void* r = base + off;
    off += (bytes + 255) & ~(size_t)255;
    return r;
  };
  __bf16* xb  = (__bf16*)alloc((size_t)MP * QD * 2);
  __bf16* Qw  = (__bf16*)alloc((size_t)2 * NH * NPAD * 64 * 2);
  __bf16* Kw  = (__bf16*)alloc((size_t)2 * NH * NPAD * 64 * 2);
  __bf16* VTw = (__bf16*)alloc((size_t)2 * NH * 48 * NPAD * 2);
  size_t zbytes = off;                         // zero xb|Q|K|VT (padding)
  __bf16* AO  = (__bf16*)alloc((size_t)MP * QD * 2);
  __bf16* wtq = (__bf16*)alloc((size_t)QD * QD * 2);
  __bf16* wtk = (__bf16*)alloc((size_t)QD * QD * 2);
  __bf16* wtv = (__bf16*)alloc((size_t)QD * QD * 2);
  __bf16* wto = (__bf16*)alloc((size_t)QD * QD * 2);
  unsigned* bits = (unsigned*)alloc((size_t)2 * WH * 4);

  hipMemsetAsync(d_ws, 0, zbytes, stream);

  k_bits<<<(2 * WH + 255) / 256, 256, 0, stream>>>(am, bits);
  k_conv_x<<<(2 * NTOK * QD + 255) / 256, 256, 0, stream>>>(x, xb);
  k_conv_wt<<<(QD * QD + 255) / 256, 256, 0, stream>>>(Wq, wtq);
  k_conv_wt<<<(QD * QD + 255) / 256, 256, 0, stream>>>(Wk, wtk);
  k_conv_wt<<<(QD * QD + 255) / 256, 256, 0, stream>>>(Wv, wtv);
  k_conv_wt<<<(QD * QD + 255) / 256, 256, 0, stream>>>(Wo, wto);

  int projWaves = 3 * (MP / 16) * (QD / 16);     // 32160
  k_proj<<<projWaves / 4, 128, 0, stream>>>(xb, wtq, wtk, wtv, Qw, Kw, VTw);

  int attnBlocks = 2 * NH * ((NPAD / 16) / 4);   // 1072
  k_attn<<<attnBlocks, 128, 0, stream>>>(Qw, Kw, VTw, bits, AO);

  int outWaves = (MP / 16) * (QD / 16);          // 10720
  k_out<<<outWaves / 4, 128, 0, stream>>>(AO, wto, bo, out);
}
